// CrossGatingBlock_19902878450436
// MI455X (gfx1250) — compile-verified, matrix operand order checked by reading, not run on hardware
//
#include <hip/hip_runtime.h>
#include <hip/hip_bf16.h>
#include <math.h>

// ---------------------------------------------------------------------------
// CDNA5 (gfx1250) CrossGatingBlock: everything is a GEMM with fused pre/epilogue.
// bf16 WMMA 16x16x32, f32 accumulate, f32 storage. wave32; one 16x16 D tile per
// wave; 4 waves per workgroup share an LDS-staged B (weight) tile held in
// per-lane fragment order (one v16bf LDS read per lane). K and the residual
// count are template constants -> fully unrolled K loop, branch-free epilogue,
// vectorized b128 stores where the D tile is contiguous in memory.
// ---------------------------------------------------------------------------

typedef __attribute__((ext_vector_type(16))) __bf16 v16bf;
typedef __attribute__((ext_vector_type(8)))  float  v8f;
typedef __attribute__((ext_vector_type(4)))  float  v4f;

#define MODE_CHAN  0
#define MODE_GRID  1
#define MODE_BLOCK 2
#define PRE_NONE 0
#define PRE_LN   1
#define PRE_MUL  2
#define ACT_NONE 0
#define ACT_RELU 1
#define ACT_GELU 2

__device__ __forceinline__ __bf16 f2bf(float f) { return (__bf16)f; }

__device__ __forceinline__ float gelu_exact(float x) {
  return 0.5f * x * (1.0f + erff(x * 0.70710678118654752f));
}

// K-dependent part of the A address (separable: kterm(kk+ko)=kterm(kk)+kterm(ko)
// because kk is a multiple of 32).
template<int MODE, int HWSHIFT>
__device__ __forceinline__ int ktermf(int k) {
  if (MODE == MODE_CHAN)      return k << HWSHIFT;
  else if (MODE == MODE_GRID) return ((k >> 4) << 10) + ((k & 15) << 3);
  else                        return ((k >> 4) << 7) + (k & 15);
}

// ---------------------------------------------------------------------------
// Per-sample LayerNorm statistics over C*H*W: stats[2n]=mean, stats[2n+1]=rstd.
// ---------------------------------------------------------------------------
__global__ __launch_bounds__(256)
void ln_stats_kernel(const float* __restrict__ src, float* __restrict__ stats,
                     int perSample) {
  __shared__ float s1[256];
  __shared__ float s2[256];
  const int n = blockIdx.x;
  const float* p = src + (size_t)n * perSample;
  float s = 0.f, ss = 0.f;
  for (int i = threadIdx.x; i < perSample; i += blockDim.x) {
    float v = p[i];
    s += v; ss += v * v;
  }
  s1[threadIdx.x] = s; s2[threadIdx.x] = ss;
  __syncthreads();
  for (int o = 128; o > 0; o >>= 1) {
    if (threadIdx.x < o) {
      s1[threadIdx.x] += s1[threadIdx.x + o];
      s2[threadIdx.x] += s2[threadIdx.x + o];
    }
    __syncthreads();
  }
  if (threadIdx.x == 0) {
    float m   = s1[0] / perSample;
    float var = s2[0] / perSample - m * m;
    stats[2 * n]     = m;
    stats[2 * n + 1] = rsqrtf(var + 1e-5f);
  }
}

// ---------------------------------------------------------------------------
// Generic fused WMMA GEMM.
//   MODE_CHAN : rows=(n,hw), K=channel, HWin = 1<<HWSHIFT. A[r,k]=in[n,k,hw].
//               UP=true maps output through the 2x ConvT tap (upoff=k*128+l).
//   MODE_GRID : rows=(n,c,p) p<64, K=g<256 (16x16 grid of 8x8 cells).
//   MODE_BLOCK: rows=(n,c,g) g<64, K=p<256 (16x16 block positions).
// B[k,d] at B + bOff + k*ldbK + d*ldbD (handles ConvT weight taps).
// Output spatial is always 128x128 (=16384), W=128.
// ---------------------------------------------------------------------------
template<int MODE, int PRE, int ACT, int HWSHIFT, bool UP, int NRES, int KTOT>
__global__ __launch_bounds__(128)
void gemm_wmma(const float* __restrict__ A, const float* __restrict__ A2,
               const float* __restrict__ B, const float* __restrict__ bias,
               const float* __restrict__ res1, const float* __restrict__ res2,
               float* __restrict__ out, const float* __restrict__ stats,
               int inStrideN, int outStrideN,
               int ldbK, int ldbD, int bOff, int upoff)
{
  // B tile in per-lane fragment order: Bs[lane*16 + e] is element e of the
  // v16bf the consuming lane needs -> one vector LDS read per lane.
  __shared__ __attribute__((aligned(32))) __bf16 Bs[512];

  const int tid   = threadIdx.x;
  const int lane  = tid & 31;
  const int laneM = lane & 15;
  const int hi    = lane >> 4;          // lane half selects K sub-range (ISA layout)
  const int row0  = (blockIdx.x * 4 + (tid >> 5)) * 16;
  const int col0  = blockIdx.y * 16;
  const int arow  = row0 + laneM;       // A row this lane supplies

  // Per-lane, k-invariant A base + LN scalars.
  size_t aBase;
  float mean = 0.f, rstd = 1.f;
  if (MODE == MODE_CHAN) {
    int n = arow >> HWSHIFT; int hw = arow & ((1 << HWSHIFT) - 1);
    aBase = (size_t)n * inStrideN + hw;
    if (PRE == PRE_LN) { mean = stats[2 * n]; rstd = stats[2 * n + 1]; }
  } else if (MODE == MODE_GRID) {
    int p = arow & 63, c = (arow >> 6) & 127, n = arow >> 13;
    aBase = (size_t)n * inStrideN + (size_t)c * 16384 + ((p >> 3) << 7) + (p & 7);
  } else {
    int g = arow & 63, c = (arow >> 6) & 127, n = arow >> 13;
    aBase = (size_t)n * inStrideN + (size_t)c * 16384 + ((g >> 3) << 11) + ((g & 7) << 4);
  }

  // Per-lane K offsets for the 16 fragment elements (ISA 16-bit A 16x32 layout:
  // element e holds K = (e<8 ? e : e+8) + 8*hi within the 32-wide K step).
  int koff[16];
#pragma unroll
  for (int e = 0; e < 16; ++e) {
    int ko = (e < 8 ? e : e + 8) + hi * 8;
    koff[e] = ktermf<MODE, HWSHIFT>(ko);
  }

  // B staging: thread tid handles lin = j*128 + tid (j=0..3); adjacent threads
  // read adjacent d (B fastest dim) -> coalesced. Precompute k-invariant parts.
  size_t sBase[4]; int sIdx[4];
#pragma unroll
  for (int j = 0; j < 4; ++j) {
    int lin = j * 128 + tid;            // 0..511
    int ls = lin & 31, es = lin >> 5;   // consuming lane, element
    int kl = (es < 8 ? es : es + 8) + (ls >> 4) * 8;
    int d  = ls & 15;
    sBase[j] = (size_t)bOff + (size_t)kl * ldbK + (size_t)(col0 + d) * ldbD;
    sIdx[j]  = ls * 16 + es;
  }

  v8f acc = {};
#pragma unroll
  for (int kk = 0; kk < KTOT; kk += 32) {
    const size_t bK = (size_t)kk * ldbK;
#pragma unroll
    for (int j = 0; j < 4; ++j)
      Bs[sIdx[j]] = f2bf(B[sBase[j] + bK]);
    __syncthreads();

    v16bf b = *(const v16bf*)(Bs + lane * 16);   // ds_load_b128 pair

    const size_t aK = aBase + (size_t)ktermf<MODE, HWSHIFT>(kk);
    v16bf a;
#pragma unroll
    for (int e = 0; e < 16; ++e) {
      float av = A[aK + koff[e]];
      if (PRE == PRE_MUL) av *= A2[aK + koff[e]];
      if (PRE == PRE_LN)  av = (av - mean) * rstd;
      a[e] = f2bf(av);
    }
    acc = __builtin_amdgcn_wmma_f32_16x16x32_bf16(
        /*neg_a=*/false, a, /*neg_b=*/false, b,
        /*c_mod=*/(short)0, acc, /*reuse_a=*/false, /*reuse_b=*/false);
    __syncthreads();
  }

  // ---- Epilogue. D layout: VGPR i holds (M = i + 8*hi, N = laneM). ----
  const int col = col0 + laneM;
  const float bv = bias[col];
  float e8[8];
#pragma unroll
  for (int i = 0; i < 8; ++i) {
    float v = acc[i] + bv;
    if (ACT == ACT_RELU) v = fmaxf(v, 0.f);
    if (ACT == ACT_GELU) v = gelu_exact(v);
    e8[i] = v;
  }

  if ((MODE == MODE_CHAN && !UP) || MODE == MODE_GRID) {
    // This lane's 8 outputs are contiguous & 16B aligned: two b128 transactions.
    const int rbase = row0 + hi * 8;
    size_t oa0;
    if (MODE == MODE_CHAN) {
      int n = rbase >> HWSHIFT; int hw = rbase & ((1 << HWSHIFT) - 1);
      oa0 = (size_t)n * outStrideN + (size_t)col * 16384 + hw;
    } else {
      int p = rbase & 63, c = (rbase >> 6) & 127, n = rbase >> 13;
      oa0 = (size_t)n * outStrideN + (size_t)c * 16384 +
            (size_t)((((col >> 4) << 3) + (p >> 3)) * 128 + ((col & 15) << 3) + (p & 7));
    }
    v4f lo = { e8[0], e8[1], e8[2], e8[3] };
    v4f hi4 = { e8[4], e8[5], e8[6], e8[7] };
    if (NRES >= 1) {
      lo  += *(const v4f*)(res1 + oa0);
      hi4 += *(const v4f*)(res1 + oa0 + 4);
    }
    if (NRES >= 2) {
      lo  += *(const v4f*)(res2 + oa0);
      hi4 += *(const v4f*)(res2 + oa0 + 4);
    }
    *(v4f*)(out + oa0)     = lo;
    *(v4f*)(out + oa0 + 4) = hi4;
  } else {
#pragma unroll
    for (int i = 0; i < 8; ++i) {
      int r = row0 + i + hi * 8;
      size_t oa;
      if (MODE == MODE_CHAN) {          // UP path (ConvT): stride-2 scatter
        int n = r >> HWSHIFT; int hw = r & ((1 << HWSHIFT) - 1);
        int hy = hw >> 6, wy = hw & 63;
        oa = (size_t)n * outStrideN + (size_t)col * 16384 +
             (size_t)(hy * 256 + wy * 2 + upoff);
      } else {                          // MODE_BLOCK: stride-16 scatter
        int g = r & 63, c = (r >> 6) & 127, n = r >> 13;
        oa = (size_t)n * outStrideN + (size_t)c * 16384 +
             (size_t)((((g >> 3) << 4) + (col >> 4)) * 128 + ((g & 7) << 4) + (col & 15));
      }
      float v = e8[i];
      if (NRES >= 1) v += res1[oa];
      if (NRES >= 2) v += res2[oa];
      out[oa] = v;
    }
  }
}

// ---------------------------------------------------------------------------
// Host orchestration
// ---------------------------------------------------------------------------
extern "C" void kernel_launch(void* const* d_in, const int* in_sizes, int n_in,
                              void* d_out, int out_size, void* d_ws, size_t ws_size,
                              hipStream_t stream) {
  (void)in_sizes; (void)n_in; (void)out_size; (void)ws_size;

  const float* x   = (const float*)d_in[0];
  const float* y   = (const float*)d_in[1];
  const float* wt  = (const float*)d_in[2];
  const float* bt  = (const float*)d_in[3];
  const float* wc1 = (const float*)d_in[4];  const float* bc1 = (const float*)d_in[5];
  const float* wc2 = (const float*)d_in[6];  const float* bc2 = (const float*)d_in[7];
  const float* wl1 = (const float*)d_in[8];  const float* bl1 = (const float*)d_in[9];
  const float* wl2 = (const float*)d_in[10]; const float* bl2 = (const float*)d_in[11];
  const float* wl3 = (const float*)d_in[12]; const float* bl3 = (const float*)d_in[13];
  const float* wl4 = (const float*)d_in[14]; const float* bl4 = (const float*)d_in[15];

  const int HW = 16384;                     // 128*128
  const size_t SZ = (size_t)8 * 128 * HW;   // one (N=8, C=128, 128x128) tensor

  float* ws    = (float*)d_ws;
  float* stats = ws;                // 4 slots * 8 samples * 2 floats
  float* xp    = ws + 256;          // shortcut_x
  float* yp    = xp + SZ;           // shortcut_y
  float* xb    = yp + SZ;           // x branch (also ConvT temp first)
  float* yb    = xb + SZ;           // y branch
  float* gx    = yb + SZ;
  float* gy    = gx + SZ;
  float* t256  = gy + SZ;           // 2*SZ: gating 2F-channel buffer
  float* t256b = t256 + 2 * SZ;     // 2*SZ: gating mixed u|v buffer

  float* outx = (float*)d_out;
  float* outy = outx + SZ;

  const dim3 blk(128);
  const int S128 = 128 * HW;        // per-sample elements, 128-channel tensor
  const int S256 = 256 * HW;        // per-sample elements, 256-channel tensor

  // --- 1) ConvT 2x2 stride 2 (+bias+ReLU): 4 taps, into xb (temp) ---
  for (int kk = 0; kk < 2; ++kk)
    for (int ll = 0; ll < 2; ++ll)
      gemm_wmma<MODE_CHAN, PRE_NONE, ACT_RELU, 12, true, 0, 256>
        <<<dim3(32768 / 64, 128 / 16), blk, 0, stream>>>(
          y, nullptr, wt, bt, nullptr, nullptr, xb, nullptr,
          /*inStrideN=*/256 * 4096, /*outStrideN=*/S128,
          /*ldbK=*/512, /*ldbD=*/4, /*bOff=*/kk * 2 + ll,
          /*upoff=*/kk * 128 + ll);

  // --- 2) x = relu(x @ wc1 + bc1) -> xp ---
  gemm_wmma<MODE_CHAN, PRE_NONE, ACT_RELU, 14, false, 0, 128>
    <<<dim3(131072 / 64, 8), blk, 0, stream>>>(
      x, nullptr, wc1, bc1, nullptr, nullptr, xp, nullptr,
      S128, S128, 128, 1, 0, 0);

  // --- 3) y = relu(convT @ wc2 + bc2) -> yp ---
  gemm_wmma<MODE_CHAN, PRE_NONE, ACT_RELU, 14, false, 0, 128>
    <<<dim3(131072 / 64, 8), blk, 0, stream>>>(
      xb, nullptr, wc2, bc2, nullptr, nullptr, yp, nullptr,
      S128, S128, 128, 1, 0, 0);

  // --- 4) xb = gelu(LN(xp) @ wl1 + bl1); yb = gelu(LN(yp) @ wl2 + bl2) ---
  ln_stats_kernel<<<8, 256, 0, stream>>>(xp, stats + 0, S128);
  gemm_wmma<MODE_CHAN, PRE_LN, ACT_GELU, 14, false, 0, 128>
    <<<dim3(131072 / 64, 8), blk, 0, stream>>>(
      xp, nullptr, wl1, bl1, nullptr, nullptr, xb, stats + 0,
      S128, S128, 128, 1, 0, 0);
  ln_stats_kernel<<<8, 256, 0, stream>>>(yp, stats + 16, S128);
  gemm_wmma<MODE_CHAN, PRE_LN, ACT_GELU, 14, false, 0, 128>
    <<<dim3(131072 / 64, 8), blk, 0, stream>>>(
      yp, nullptr, wl2, bl2, nullptr, nullptr, yb, stats + 16,
      S128, S128, 128, 1, 0, 0);

  // --- 5) Spatial gating weights for each branch ---
  for (int b = 0; b < 2; ++b) {
    const float* w1 = (const float*)d_in[16 + b * 8];
    const float* b1 = (const float*)d_in[17 + b * 8];
    const float* w2 = (const float*)d_in[18 + b * 8];
    const float* b2 = (const float*)d_in[19 + b * 8];
    const float* w3 = (const float*)d_in[20 + b * 8];
    const float* b3 = (const float*)d_in[21 + b * 8];
    const float* w4 = (const float*)d_in[22 + b * 8];
    const float* b4 = (const float*)d_in[23 + b * 8];
    const float* brIn = b ? yb : xb;
    float*       gout = b ? gy : gx;
    float*       st   = stats + 32 + b * 16;

    ln_stats_kernel<<<8, 256, 0, stream>>>(brIn, st, S128);
    // t256 = gelu(LN(in) @ w1 + b1)  [128 -> 256 channels]
    gemm_wmma<MODE_CHAN, PRE_LN, ACT_GELU, 14, false, 0, 128>
      <<<dim3(131072 / 64, 16), blk, 0, stream>>>(
        brIn, nullptr, w1, b1, nullptr, nullptr, t256, st,
        S128, S256, 256, 1, 0, 0);
    // grid MLP on u (channels 0..127): mix across 256 grid cells
    gemm_wmma<MODE_GRID, PRE_NONE, ACT_NONE, 14, false, 0, 256>
      <<<dim3(65536 / 64, 16), blk, 0, stream>>>(
        t256, nullptr, w2, b2, nullptr, nullptr, t256b, nullptr,
        S256, S256, 256, 1, 0, 0);
    // block MLP on v (channels 128..255): mix across 256 block positions
    gemm_wmma<MODE_BLOCK, PRE_NONE, ACT_NONE, 14, false, 0, 256>
      <<<dim3(65536 / 64, 16), blk, 0, stream>>>(
        t256 + (size_t)128 * HW, nullptr, w3, b3, nullptr, nullptr,
        t256b + (size_t)128 * HW, nullptr,
        S256, S256, 256, 1, 0, 0);
    // gate = [u'|v'] @ w4 + b4  [256 -> 128 channels]
    gemm_wmma<MODE_CHAN, PRE_NONE, ACT_NONE, 14, false, 0, 256>
      <<<dim3(131072 / 64, 8), blk, 0, stream>>>(
        t256b, nullptr, w4, b4, nullptr, nullptr, gout, nullptr,
        S256, S128, 128, 1, 0, 0);
  }

  // --- 6) y_out = (yb * gx) @ wl3 + bl3 + yp ---
  gemm_wmma<MODE_CHAN, PRE_MUL, ACT_NONE, 14, false, 1, 128>
    <<<dim3(131072 / 64, 8), blk, 0, stream>>>(
      yb, gx, wl3, bl3, yp, nullptr, outy, nullptr,
      S128, S128, 128, 1, 0, 0);

  // --- 7) x_out = (xb * gy) @ wl4 + bl4 + y_out + xp ---
  gemm_wmma<MODE_CHAN, PRE_MUL, ACT_NONE, 14, false, 2, 128>
    <<<dim3(131072 / 64, 8), blk, 0, stream>>>(
      xb, gy, wl4, bl4, outy, xp, outx, nullptr,
      S128, S128, 128, 1, 0, 0);
}
